// LSTMAutoEncoder_51462298141063
// MI455X (gfx1250) — matrix operand order repaired and checked by reading, not compiled, not run on hardware
//
#include <hip/hip_runtime.h>
#include <hip/hip_bf16.h>

typedef _Float16 half_t;
typedef __attribute__((ext_vector_type(16))) _Float16 v16h;
typedef __attribute__((ext_vector_type(8)))  float    v8f;

constexpr int Bsz = 512, Tsz = 128, Fsz = 64, Hsz = 512, Ssz = 128;

// ---------- fragment loaders (CDNA5 16-bit WMMA layouts, wave32) ----------
// Pointers arrive pre-offset for this lane; only the K offset varies per call.
// A 16x32 (MxK): halves[0..7] at p, halves[8..15] at p+16 (K-split by lane half).
__device__ __forceinline__ v16h ldfragA(const half_t* p) {
    union { v16h v; uint4 q[2]; } u;
    u.q[0] = *(const uint4*)(p);
    u.q[1] = *(const uint4*)(p + 16);
    return u.v;
}
// B 32x16 (KxN): halves[0..7] at p, halves[8..15] at p+16 of weight row n (K-split by lane half).
__device__ __forceinline__ v16h ldfragB(const half_t* p) {
    union { v16h v; uint4 q[2]; } u;
    u.q[0] = *(const uint4*)(p);
    u.q[1] = *(const uint4*)(p + 16);
    return u.v;
}

__device__ __forceinline__ float fast_sigmoid(float x) {
    return __builtin_amdgcn_rcpf(1.0f + __expf(-x));
}
__device__ __forceinline__ float fast_tanh(float x) {
#if __has_builtin(__builtin_amdgcn_tanhf)
    return __builtin_amdgcn_tanhf(x);
#else
    float e = __expf(2.0f * x);
    return 1.0f - 2.0f * __builtin_amdgcn_rcpf(e + 1.0f);
#endif
}

// ---------- one LSTM cell step for a 16-row batch tile ----------
// Wave w owns hidden columns [w*32, w*32+32): computes the 4 matching gate
// tiles (i/f/g/o x 2 n-tiles) so the gate elementwise stays in registers.
// hb is LDS [16][512] f16, updated in place (two barriers).
__device__ __forceinline__ void lstm_step(
    const half_t* xsrc, int xrs, int kx,       // input activations [16][kx]
    const half_t* __restrict__ wih,            // [2048][kx] row-major
    half_t* hb,                                // LDS h state [16][512]
    const half_t* __restrict__ whh,            // [2048][512] row-major
    const float*  __restrict__ bias,           // [2048]
    float (&cst)[2][8])                        // cell state, reg-resident
{
    const int lane = threadIdx.x & 31;
    const int ln   = lane & 15;
    const int hi   = lane >> 4;
    const int j0   = (threadIdx.x >> 5) * 32;

    v8f acc[8];
#pragma unroll
    for (int g = 0; g < 4; ++g)
#pragma unroll
        for (int nt = 0; nt < 2; ++nt) {
            float bv = bias[g * Hsz + j0 + nt * 16 + ln];
#pragma unroll
            for (int r = 0; r < 8; ++r) acc[g * 2 + nt][r] = bv;
        }

    // ---- x @ Wih^T ----  (rolled loop: 1 A frag + 8 B frags live, no spills)
    {
        const half_t* ax = xsrc + (size_t)ln * xrs + hi * 8;
        const half_t* bx[8];
#pragma unroll
        for (int g = 0; g < 4; ++g)
#pragma unroll
            for (int nt = 0; nt < 2; ++nt)
                bx[g * 2 + nt] = wih + (size_t)(g * Hsz + j0 + nt * 16 + ln) * kx + hi * 8;

#pragma unroll 1
        for (int kt = 0; kt < kx; kt += 32) {
            v16h a = ldfragA(ax + kt);
#pragma unroll
            for (int i = 0; i < 8; ++i) {
                v16h b = ldfragB(bx[i] + kt);
                acc[i] = __builtin_amdgcn_wmma_f32_16x16x32_f16(
                    false, a, false, b, (short)0, acc[i], false, false);
            }
        }
    }
    // ---- h @ Whh^T ----
    {
        const half_t* ah = hb + ln * Hsz + hi * 8;
        const half_t* bh[8];
#pragma unroll
        for (int g = 0; g < 4; ++g)
#pragma unroll
            for (int nt = 0; nt < 2; ++nt)
                bh[g * 2 + nt] = whh + (size_t)(g * Hsz + j0 + nt * 16 + ln) * Hsz + hi * 8;

#pragma unroll 1
        for (int kt = 0; kt < Hsz; kt += 32) {
            v16h a = ldfragA(ah + kt);
#pragma unroll
            for (int i = 0; i < 8; ++i) {
                v16h b = ldfragB(bh[i] + kt);
                acc[i] = __builtin_amdgcn_wmma_f32_16x16x32_f16(
                    false, a, false, b, (short)0, acc[i], false, false);
            }
        }
    }
    __syncthreads();   // all waves done reading old h

    // gate elementwise in registers; C-layout: m = r + 8*(lane>=16), n = lane%16
#pragma unroll
    for (int nt = 0; nt < 2; ++nt) {
        const int n     = j0 + nt * 16 + ln;
        const int mbase = hi * 8;
#pragma unroll
        for (int r = 0; r < 8; ++r) {
            float iv = acc[0 + nt][r];
            float fv = acc[2 + nt][r];
            float gv = acc[4 + nt][r];
            float ov = acc[6 + nt][r];
            float cn = fast_sigmoid(fv) * cst[nt][r] + fast_sigmoid(iv) * fast_tanh(gv);
            cst[nt][r] = cn;
            float hn = fast_sigmoid(ov) * fast_tanh(cn);
            hb[(mbase + r) * Hsz + n] = (half_t)hn;
        }
    }
    __syncthreads();   // new h visible before next consumer
}

// ---------- persistent autoencoder kernel: one WG per 16 batch rows ----------
__global__ __launch_bounds__(512, 1) void lstm_ae_kernel(
    const half_t* __restrict__ xh,
    const half_t* __restrict__ ewih0, const half_t* __restrict__ ewhh0, const float* __restrict__ eb0,
    const half_t* __restrict__ ewih1, const half_t* __restrict__ ewhh1, const float* __restrict__ eb1,
    const half_t* __restrict__ dwih0, const half_t* __restrict__ dwhh0, const float* __restrict__ db0,
    const half_t* __restrict__ dwih1, const half_t* __restrict__ dwhh1, const float* __restrict__ db1,
    const half_t* __restrict__ linw,  const float* __restrict__ linb,
    float* __restrict__ out)
{
    __shared__ half_t hb0[16 * Hsz];   // 16 KB
    __shared__ half_t hb1[16 * Hsz];   // 16 KB
    __shared__ half_t yb [16 * Fsz];   // 2 KB decoder feedback

    const int b0 = blockIdx.x * 16;

    for (int i = threadIdx.x; i < 16 * Hsz; i += 512) { hb0[i] = (half_t)0.f; hb1[i] = (half_t)0.f; }
    __syncthreads();

    float c0[2][8], c1[2][8];
#pragma unroll
    for (int nt = 0; nt < 2; ++nt)
#pragma unroll
        for (int r = 0; r < 8; ++r) { c0[nt][r] = 0.f; c1[nt][r] = 0.f; }

    // ---------------- encoder ----------------
#pragma unroll 1
    for (int t = 0; t < Tsz; ++t) {
        const half_t* xsrc = xh + (size_t)b0 * Tsz * Fsz + (size_t)t * Fsz;
        lstm_step(xsrc, Tsz * Fsz, Fsz, ewih0, hb0, ewhh0, eb0, c0);
        lstm_step(hb0,  Hsz,       Hsz, ewih1, hb1, ewhh1, eb1, c1);
    }

    // seed decoder input with x[:, T-1, :]
    for (int i = threadIdx.x; i < 16 * Fsz; i += 512) {
        int m = i >> 6, f = i & 63;
        yb[i] = xh[(size_t)(b0 + m) * Tsz * Fsz + (size_t)(Tsz - 1) * Fsz + f];
    }
    __syncthreads();

    const int lane = threadIdx.x & 31;
    const int ln   = lane & 15;
    const int hi   = lane >> 4;
    const int w    = threadIdx.x >> 5;

    // ---------------- decoder ----------------
#pragma unroll 1
    for (int s = 0; s < Ssz; ++s) {
        lstm_step(yb,  Fsz, Fsz, dwih0, hb0, dwhh0, db0, c0);
        lstm_step(hb0, Hsz, Hsz, dwih1, hb1, dwhh1, db1, c1);

        // linear read-out: y = h1 @ lin_W^T + lin_b  (16x512 @ 512x64)
        if (w < 4) {
            const int n0 = w * 16;
            v8f acc;
            {
                float bv = linb[n0 + ln];
#pragma unroll
                for (int r = 0; r < 8; ++r) acc[r] = bv;
            }
            const half_t* ah = hb1 + ln * Hsz + hi * 8;
            const half_t* bw = linw + (size_t)(n0 + ln) * Hsz + hi * 8;
#pragma unroll 1
            for (int kt = 0; kt < Hsz; kt += 32) {
                v16h a = ldfragA(ah + kt);
                v16h b = ldfragB(bw + kt);
                acc = __builtin_amdgcn_wmma_f32_16x16x32_f16(
                    false, a, false, b, (short)0, acc, false, false);
            }
            const int n     = n0 + ln;
            const int mbase = hi * 8;
#pragma unroll
            for (int r = 0; r < 8; ++r) {
                int m = mbase + r;
                // out[b][S-1-s][f]  (reference flips the step axis)
                out[(size_t)(b0 + m) * Ssz * Fsz + (size_t)(Ssz - 1 - s) * Fsz + n] = acc[r];
                yb[m * Fsz + n] = (half_t)acc[r];
            }
        }
        __syncthreads();   // yb ready for next step's layer-0
    }
}

// ---------- f32 -> f16 conversion ----------
__global__ void f32_to_f16_kernel(const float* __restrict__ src, half_t* __restrict__ dst, int n) {
    int i = blockIdx.x * blockDim.x + threadIdx.x;
    if (i < n) dst[i] = (half_t)src[i];
}

extern "C" void kernel_launch(void* const* d_in, const int* in_sizes, int n_in,
                              void* d_out, int out_size, void* d_ws, size_t ws_size,
                              hipStream_t stream) {
    (void)in_sizes; (void)n_in; (void)out_size; (void)ws_size;

    const float* input_seq = (const float*)d_in[0];
    // d_in[1] = step_window (== Ssz, compile-time here)
    const float* eWih0 = (const float*)d_in[2];
    const float* eWhh0 = (const float*)d_in[3];
    const float* eb0   = (const float*)d_in[4];
    const float* eWih1 = (const float*)d_in[5];
    const float* eWhh1 = (const float*)d_in[6];
    const float* eb1   = (const float*)d_in[7];
    const float* dWih0 = (const float*)d_in[8];
    const float* dWhh0 = (const float*)d_in[9];
    const float* db0   = (const float*)d_in[10];
    const float* dWih1 = (const float*)d_in[11];
    const float* dWhh1 = (const float*)d_in[12];
    const float* db1   = (const float*)d_in[13];
    const float* linW  = (const float*)d_in[14];
    const float* linb  = (const float*)d_in[15];

    half_t* ws = (half_t*)d_ws;
    const size_t nX    = (size_t)Bsz * Tsz * Fsz;        // 4,194,304
    const size_t nWihF = (size_t)4 * Hsz * Fsz;          //   131,072
    const size_t nWH   = (size_t)4 * Hsz * Hsz;          // 1,048,576
    const size_t nLin  = (size_t)Fsz * Hsz;              //    32,768

    half_t* xh     = ws;                size_t o = nX;
    half_t* ewih0h = ws + o; o += nWihF;
    half_t* ewhh0h = ws + o; o += nWH;
    half_t* ewih1h = ws + o; o += nWH;
    half_t* ewhh1h = ws + o; o += nWH;
    half_t* dwih0h = ws + o; o += nWihF;
    half_t* dwhh0h = ws + o; o += nWH;
    half_t* dwih1h = ws + o; o += nWH;
    half_t* dwhh1h = ws + o; o += nWH;
    half_t* linwh  = ws + o; o += nLin;

    auto conv = [&](const float* s, half_t* d, size_t n) {
        f32_to_f16_kernel<<<(unsigned)((n + 255) / 256), 256, 0, stream>>>(s, d, (int)n);
    };
    conv(input_seq, xh, nX);
    conv(eWih0, ewih0h, nWihF);
    conv(eWhh0, ewhh0h, nWH);
    conv(eWih1, ewih1h, nWH);
    conv(eWhh1, ewhh1h, nWH);
    conv(dWih0, dwih0h, nWihF);
    conv(dWhh0, dwhh0h, nWH);
    conv(dWih1, dwih1h, nWH);
    conv(dWhh1, dwhh1h, nWH);
    conv(linW,  linwh,  nLin);

    lstm_ae_kernel<<<Bsz / 16, 512, 0, stream>>>(
        xh,
        ewih0h, ewhh0h, eb0,
        ewih1h, ewhh1h, eb1,
        dwih0h, dwhh0h, db0,
        dwih1h, dwhh1h, db1,
        linwh, linb,
        (float*)d_out);
}
